// GenerativeModel_82420422410826
// MI455X (gfx1250) — matrix operand order repaired and checked by reading, not compiled, not run on hardware
//
#include <hip/hip_runtime.h>
#include <cstdint>

typedef __attribute__((ext_vector_type(16))) _Float16 v16h;
typedef __attribute__((ext_vector_type(8)))  _Float16 v8h;
typedef __attribute__((ext_vector_type(8)))  float    v8f;
typedef __attribute__((ext_vector_type(4)))  int      v4i;

#define TFULL   65536
#define CCH     256
#define NLAYER  16
#define TILE_N  64
#define LDS_STRIDE 520   // 512 K-halves + 8 pad (bank-conflict mitigation)

#if __has_builtin(__builtin_amdgcn_global_load_async_to_lds_b128)
#define HAS_ASYNC_LDS 1
typedef __attribute__((address_space(1))) v4i* gptr128_t;  // global int4*
typedef __attribute__((address_space(3))) v4i* lptr128_t;  // LDS int4*
#else
#define HAS_ASYNC_LDS 0
#endif

// ---------------------------------------------------------------------------
// x [C][T] f32 (channel-major) -> X [T][C] f16 (time-major), LDS-tiled transpose
// ---------------------------------------------------------------------------
__global__ __launch_bounds__(256)
void xpose_f32_to_f16(const float* __restrict__ in, _Float16* __restrict__ out) {
  __shared__ float tile[32][33];
  const int t0 = blockIdx.x * 32;
  const int c0 = blockIdx.y * 32;
  const int tx = threadIdx.x;            // 0..31
  for (int j = threadIdx.y; j < 32; j += 8)
    tile[j][tx] = in[(size_t)(c0 + j) * TFULL + t0 + tx];   // coalesced over t
  __syncthreads();
  for (int j = threadIdx.y; j < 32; j += 8)
    out[(size_t)(t0 + j) * CCH + c0 + tx] = (_Float16)tile[tx][j]; // coalesced over c
}

// ---------------------------------------------------------------------------
// W [l][o][i][k] f32 -> Wh [l][o][k*256+i] f16  (row-major 256x512 per layer)
// flat output index == flat iteration index (i fastest, then k, o, l)
// ---------------------------------------------------------------------------
__global__ __launch_bounds__(256)
void wcvt(const float* __restrict__ W, _Float16* __restrict__ Wh) {
  const int idx = blockIdx.x * 256 + threadIdx.x;
  const int i = idx & 255;
  const int k = (idx >> 8) & 1;
  const int o = (idx >> 9) & 255;
  const int l = idx >> 17;
  Wh[idx] = (_Float16)W[(((size_t)l * 256 + o) * 256 + i) * 2 + k];
}

// ---------------------------------------------------------------------------
// One dilated-conv layer as a 256x512 x 512xN GEMM + bias + tanh.
// Block: 256 threads (8 waves). Output tile: M=256 x N=64 time steps.
// Waves: 4 along M (64 rows each) x 2 along N (32 cols each).
// ---------------------------------------------------------------------------
__global__ __launch_bounds__(256)
void wavenet_layer(const _Float16* __restrict__ Xin,
                   _Float16* __restrict__ Xout,
                   const _Float16* __restrict__ Wl,   // 256x512 row-major f16
                   const float*    __restrict__ bl,   // 256 bias
                   float*          __restrict__ fout, // non-null on final layer
                   int Tin, int Tout, int dil) {
  __shared__ _Float16 ldsB[TILE_N * LDS_STRIDE];
  const int tid   = threadIdx.x;
  const int tileT = blockIdx.x * TILE_N;

  // Stage B tile into LDS: 64 columns, each column = concat(X[t], X[t+dil]) (512 halves).
  // 4096 16-byte chunks / 256 threads = 16 chunks each. Use the CDNA5 async
  // global->LDS DMA path (ASYNCcnt) when the toolchain exposes it.
  #pragma unroll
  for (int j = 0; j < 16; ++j) {
    const int chunk = tid + j * 256;        // 0..4095
    const int col   = chunk >> 6;           // 0..63
    const int kOff  = (chunk & 63) * 8;     // 0..504 (halves)
    const int tap   = kOff >> 8;            // 0 or 1
    const int t     = tileT + col + tap * dil;
    _Float16* lp = &ldsB[col * LDS_STRIDE + kOff];
    if (t < Tin) {
      const _Float16* gp = Xin + (size_t)t * CCH + (kOff & 255);
#if HAS_ASYNC_LDS
      // generic LDS pointer keeps the byte offset in its low 32 bits;
      // route through integers to form the addrspace-qualified operands.
      __builtin_amdgcn_global_load_async_to_lds_b128(
          (gptr128_t)(uintptr_t)gp,
          (lptr128_t)(unsigned)(uintptr_t)lp,
          0, 0);
#else
      *(v8h*)lp = *(const v8h*)gp;
#endif
    } else {
      v8h z = {};
      *(v8h*)lp = z;                         // zero-fill tail columns
    }
  }
#if HAS_ASYNC_LDS
#if __has_builtin(__builtin_amdgcn_s_wait_asynccnt)
  __builtin_amdgcn_s_wait_asynccnt(0);
#else
  asm volatile("s_wait_asynccnt 0" ::: "memory");
#endif
#endif
  __syncthreads();

  const int lane  = tid & 31;
  const int wave  = tid >> 5;
  const int waveM = wave & 3;   // 0..3 -> M offset waveM*64
  const int waveN = wave >> 2;  // 0..1 -> N offset waveN*32
  const int r     = lane & 15;
  const int hf    = lane >> 4;  // lane half

  v8f acc[4][2] = {};

  for (int kk = 0; kk < 512; kk += 32) {
    // B fragments (32x16 f16): per lane two 8-half chunks at K {hf*8, 16+hf*8}.
    v16h bf[2];
    #pragma unroll
    for (int nt = 0; nt < 2; ++nt) {
      const _Float16* p =
          &ldsB[(waveN * 32 + nt * 16 + r) * LDS_STRIDE + kk + hf * 8];
      v8h lo = *(const v8h*)p;
      v8h hi = *(const v8h*)(p + 16);
      bf[nt] = __builtin_shufflevector(lo, hi,
                 0,1,2,3,4,5,6,7,8,9,10,11,12,13,14,15);
    }
    // A fragments (16x32 f16): per lane two 8-half chunks at K {hf*8, 16+hf*8}.
    #pragma unroll
    for (int mt = 0; mt < 4; ++mt) {
      const _Float16* p =
          Wl + (size_t)(waveM * 64 + mt * 16 + r) * 512 + kk + hf * 8;
      v8h lo = *(const v8h*)p;
      v8h hi = *(const v8h*)(p + 16);
      v16h af = __builtin_shufflevector(lo, hi,
                  0,1,2,3,4,5,6,7,8,9,10,11,12,13,14,15);
      #pragma unroll
      for (int nt = 0; nt < 2; ++nt)
        acc[mt][nt] = __builtin_amdgcn_wmma_f32_16x16x32_f16(
            false, af, false, bf[nt], (short)0, acc[mt][nt], false, false);
    }
  }

  // Epilogue: bias + tanh, pack 8 consecutive rows -> one 16B f16 store.
  #pragma unroll
  for (int mt = 0; mt < 4; ++mt) {
    const int rowBase = waveM * 64 + mt * 16 + hf * 8; // D layout: M = v + 8*hf
    #pragma unroll
    for (int nt = 0; nt < 2; ++nt) {
      const int t = tileT + waveN * 32 + nt * 16 + r;  // D layout: N = lane&15
      if (t < Tout) {
        v8h hv;
        float f[8];
        #pragma unroll
        for (int v = 0; v < 8; ++v) {
          const float y = tanhf(acc[mt][nt][v] + bl[rowBase + v]);
          f[v]  = y;
          hv[v] = (_Float16)y;
        }
        if (fout) {                          // final layer: Tout==1 -> t==0
          #pragma unroll
          for (int v = 0; v < 8; ++v) fout[rowBase + v] = f[v];
        } else {
          *(v8h*)(Xout + (size_t)t * CCH + rowBase) = hv;
        }
      }
    }
  }
}

// ---------------------------------------------------------------------------
extern "C" void kernel_launch(void* const* d_in, const int* in_sizes, int n_in,
                              void* d_out, int out_size, void* d_ws, size_t ws_size,
                              hipStream_t stream) {
  (void)in_sizes; (void)n_in; (void)out_size; (void)ws_size;
  const float* x = (const float*)d_in[0];   // (1, 256, 65536)
  const float* W = (const float*)d_in[1];   // (16, 256, 256, 2)
  const float* b = (const float*)d_in[2];   // (16, 256)

  char* ws = (char*)d_ws;
  _Float16* Wh = (_Float16*)ws;                                    // 4 MB
  _Float16* X0 = (_Float16*)(ws + (size_t)NLAYER * CCH * 512 * 2); // 32 MB
  _Float16* X1 = X0 + (size_t)TFULL * CCH;                         // 32 MB

  wcvt<<<(NLAYER * CCH * 512) / 256, 256, 0, stream>>>(W, Wh);
  xpose_f32_to_f16<<<dim3(TFULL / 32, CCH / 32), dim3(32, 8), 0, stream>>>(x, X0);

  _Float16* cur = X0;
  _Float16* nxt = X1;
  int Tin = TFULL;
  for (int l = 0; l < NLAYER; ++l) {
    const int dil  = 1 << l;
    const int Tout = Tin - dil;
    const int blocks = (Tout + TILE_N - 1) / TILE_N;
    float* fout = (l == NLAYER - 1) ? (float*)d_out : nullptr;
    wavenet_layer<<<blocks, 256, 0, stream>>>(
        cur, nxt, Wh + (size_t)l * CCH * 512, b + (size_t)l * CCH,
        fout, Tin, Tout, dil);
    _Float16* tmp = cur; cur = nxt; nxt = tmp;
    Tin = Tout;
  }
}
